// HomoMusicGNN_83829171683609
// MI455X (gfx1250) — matrix operand order repaired and checked by reading, not compiled, not run on hardware
//
#include <hip/hip_runtime.h>

// ---- problem constants (match reference) ----
#define NO_  80000
#define NC_  20000
#define NS_  5000
#define NN_  500
#define ND_  50
#define NTOT (NO_ + NC_ + NS_ + NN_ + ND_)   // 105550
#define MPAD 105552                          // NTOT rounded up to 16
#define HDIM 256
#define NCLS 1024
#define NLAYER 3
#define LN_EPS 1e-5f

#define O_OCC   0
#define O_CHORD 80000
#define O_SEC   100000
#define O_NOTE  105000
#define O_SD    105500

// LDS row stride for staged B strips: 256 + 8 halfs. Row-to-row delta =
// 528B = 132 dwords -> adjacent fragment lanes land 4 banks apart, so a
// 16-lane x b128 read covers all 64 LDS banks exactly once (no conflicts).
#define LDSTRIDE 264

typedef __attribute__((ext_vector_type(16))) _Float16 v16h;
typedef __attribute__((ext_vector_type(8)))  _Float16 v8h;
typedef __attribute__((ext_vector_type(8)))  float    v8f;

// ---------------- WMMA fragment helpers (CDNA5 16x16x32 f16, wave32) -------
// A: 16x32 (MxK) row-major source. Layout: lanes 0-15 M=0..15 (laneHi adds
// +16 to K base); element s -> K=16*laneHi+s (contiguous, matches B).
// Per lane this is two contiguous 16B chunks -> compiler emits 2x b128 loads.
__device__ __forceinline__ v16h load_a_frag(const _Float16* __restrict__ A,
                                            int lda, int m0, int k0, int lane) {
  const int laneLo = lane & 15;
  const int laneHi = lane >> 4;
  const _Float16* row = A + (size_t)(m0 + laneLo) * lda + k0 + 16 * laneHi;
  v16h a;
#pragma unroll
  for (int v = 0; v < 8; ++v) {
    const int k = v << 1;  // 2*v
    a[2 * v]     = row[k];
    a[2 * v + 1] = row[k + 1];
  }
  return a;
}

// B fragment from an LDS-staged W^T strip: lane holds column n = lane&15 of
// the tile; 16 contiguous K halfs starting at 16*laneHi. Loaded as two 16B
// chunks (ds_load_b128) to avoid any 32B alignment assumption.
__device__ __forceinline__ v16h lds_b_frag(const _Float16* p) {
  const v8h* q = (const v8h*)p;
  v8h lo = q[0], hi = q[1];
  v16h r;
#pragma unroll
  for (int i = 0; i < 8; ++i) { r[i] = lo[i]; r[8 + i] = hi[i]; }
  return r;
}

// ---------------- GEMM kernels ---------------------------------------------
// Block = 8 waves sharing one 64-col weight strip (staged in LDS once);
// wave w handles mTile = mBlock*8 + w, computing a 16x64 strip with 4
// accumulators. B reads come from LDS (low latency, no VMEM pressure);
// only the A fragments (2x b128) hit global memory per k-step.
// D = A1@B1^T + A2@B2^T + bias ; K = HDIM, Nout multiple of 64.
__global__ __launch_bounds__(256) void gemm_dual_wmma(
    const _Float16* __restrict__ A1, const _Float16* __restrict__ B1,
    const _Float16* __restrict__ A2, const _Float16* __restrict__ B2,
    const float* __restrict__ bias, float* __restrict__ D,
    int Mtiles, int Nout, int K) {
  __shared__ _Float16 lB[64 * LDSTRIDE];
  const int nStrips = Nout >> 6;
  const int mBlock = blockIdx.x / nStrips;
  const int n0 = (blockIdx.x - mBlock * nStrips) << 6;
  const int lane = threadIdx.x & 31;
  const int laneLo = lane & 15;
  const int laneHi = lane >> 4;
  const int mTile = mBlock * 8 + (threadIdx.x >> 5);
  const bool valid = (mTile < Mtiles);        // wave-uniform
  v8f acc0 = {}, acc1 = {}, acc2 = {}, acc3 = {};
#pragma unroll
  for (int ph = 0; ph < 2; ++ph) {
    const _Float16* Bsrc = ph ? B2 : B1;
    const _Float16* Asrc = ph ? A2 : A1;
    if (ph) __syncthreads();                  // prior-phase B reads complete
    {   // cooperative stage: 64 rows x 512B, 4 threads/row x 8 uint4 each
      const int row = threadIdx.x >> 2;
      const int cb = (threadIdx.x & 3) * 8;
      const uint4* g = (const uint4*)(Bsrc + (size_t)(n0 + row) * K);
      uint4* s = (uint4*)(lB + row * LDSTRIDE);
#pragma unroll
      for (int i = 0; i < 8; ++i) s[cb + i] = g[cb + i];
    }
    __syncthreads();
    if (valid) {
      for (int k0 = 0; k0 < K; k0 += 32) {
        v16h a = load_a_frag(Asrc, K, mTile * 16, k0, lane);
        const _Float16* lb = lB + (size_t)laneLo * LDSTRIDE + k0 + 16 * laneHi;
        v16h p0 = lds_b_frag(lb);
        v16h p1 = lds_b_frag(lb + 16 * LDSTRIDE);
        v16h p2 = lds_b_frag(lb + 32 * LDSTRIDE);
        v16h p3 = lds_b_frag(lb + 48 * LDSTRIDE);
        acc0 = __builtin_amdgcn_wmma_f32_16x16x32_f16(false, a, false, p0,
                                                      (short)0, acc0,
                                                      false, false);
        acc1 = __builtin_amdgcn_wmma_f32_16x16x32_f16(false, a, false, p1,
                                                      (short)0, acc1,
                                                      false, false);
        acc2 = __builtin_amdgcn_wmma_f32_16x16x32_f16(false, a, false, p2,
                                                      (short)0, acc2,
                                                      false, false);
        acc3 = __builtin_amdgcn_wmma_f32_16x16x32_f16(false, a, false, p3,
                                                      (short)0, acc3,
                                                      false, false);
      }
    }
  }
  if (!valid) return;
  const v8f accs[4] = {acc0, acc1, acc2, acc3};
#pragma unroll
  for (int j = 0; j < 4; ++j) {
    const int n = n0 + 16 * j + laneLo;
    const float bv = bias[n];
#pragma unroll
    for (int e = 0; e < 8; ++e) {
      const int m = mTile * 16 + 8 * laneHi + e;
      D[(size_t)m * Nout + n] = accs[j][e] + bv;
    }
  }
}

// D = A@B^T + bias ; same LDS-staged structure, single matrix product.
__global__ __launch_bounds__(256) void gemm_wmma(
    const _Float16* __restrict__ A, const _Float16* __restrict__ B,
    const float* __restrict__ bias, float* __restrict__ D,
    int Mtiles, int Nout, int K) {
  __shared__ _Float16 lB[64 * LDSTRIDE];
  const int nStrips = Nout >> 6;
  const int mBlock = blockIdx.x / nStrips;
  const int n0 = (blockIdx.x - mBlock * nStrips) << 6;
  const int lane = threadIdx.x & 31;
  const int laneLo = lane & 15;
  const int laneHi = lane >> 4;
  const int mTile = mBlock * 8 + (threadIdx.x >> 5);
  const bool valid = (mTile < Mtiles);
  {
    const int row = threadIdx.x >> 2;
    const int cb = (threadIdx.x & 3) * 8;
    const uint4* g = (const uint4*)(B + (size_t)(n0 + row) * K);
    uint4* s = (uint4*)(lB + row * LDSTRIDE);
#pragma unroll
    for (int i = 0; i < 8; ++i) s[cb + i] = g[cb + i];
  }
  __syncthreads();
  if (!valid) return;
  v8f acc0 = {}, acc1 = {}, acc2 = {}, acc3 = {};
  for (int k0 = 0; k0 < K; k0 += 32) {
    v16h a = load_a_frag(A, K, mTile * 16, k0, lane);
    const _Float16* lb = lB + (size_t)laneLo * LDSTRIDE + k0 + 16 * laneHi;
    v16h p0 = lds_b_frag(lb);
    v16h p1 = lds_b_frag(lb + 16 * LDSTRIDE);
    v16h p2 = lds_b_frag(lb + 32 * LDSTRIDE);
    v16h p3 = lds_b_frag(lb + 48 * LDSTRIDE);
    acc0 = __builtin_amdgcn_wmma_f32_16x16x32_f16(false, a, false, p0,
                                                  (short)0, acc0, false, false);
    acc1 = __builtin_amdgcn_wmma_f32_16x16x32_f16(false, a, false, p1,
                                                  (short)0, acc1, false, false);
    acc2 = __builtin_amdgcn_wmma_f32_16x16x32_f16(false, a, false, p2,
                                                  (short)0, acc2, false, false);
    acc3 = __builtin_amdgcn_wmma_f32_16x16x32_f16(false, a, false, p3,
                                                  (short)0, acc3, false, false);
  }
  const v8f accs[4] = {acc0, acc1, acc2, acc3};
#pragma unroll
  for (int j = 0; j < 4; ++j) {
    const int n = n0 + 16 * j + laneLo;
    const float bv = bias[n];
#pragma unroll
    for (int e = 0; e < 8; ++e) {
      const int m = mTile * 16 + 8 * laneHi + e;
      D[(size_t)m * Nout + n] = accs[j][e] + bv;
    }
  }
}

// ---------------- graph / elementwise kernels ------------------------------
__global__ void scatter_chord(const float* __restrict__ x_chord,
                              const int* __restrict__ ei, int nE,
                              float* __restrict__ cpo) {
  int e = blockIdx.x * blockDim.x + threadIdx.x;
  if (e >= nE) return;
  int s = ei[e];
  int d = ei[nE + e];
  const float4* src = (const float4*)(x_chord + (size_t)s * 32);
  float4* dst = (float4*)(cpo + (size_t)d * 32);
#pragma unroll
  for (int i = 0; i < 8; ++i) dst[i] = src[i];
}

__global__ void count_deg(const int* __restrict__ ei, int nE, int dstOff,
                          float* __restrict__ cnt) {
  int e = blockIdx.x * blockDim.x + threadIdx.x;
  if (e >= nE) return;
  unsafeAtomicAdd(&cnt[ei[nE + e] + dstOff], 1.0f);
}

__global__ void inv_count(float* __restrict__ cnt) {
  int i = blockIdx.x * blockDim.x + threadIdx.x;
  if (i >= NTOT) return;
  cnt[i] = 1.0f / fmaxf(cnt[i], 1.0f);
}

// agg[dst] += h[src] ; 64 threads per edge, 4 floats each
__global__ void scatter_add(const float* __restrict__ h,
                            const int* __restrict__ ei, int nE,
                            int srcOff, int dstOff, float* __restrict__ agg) {
  size_t t = (size_t)blockIdx.x * blockDim.x + threadIdx.x;
  int e = (int)(t >> 6);
  if (e >= nE) return;
  int c = ((int)t & 63) * 4;
  int s = ei[e] + srcOff;
  int d = ei[nE + e] + dstOff;
  const float* hs = h + (size_t)s * HDIM + c;
  float* ad = agg + (size_t)d * HDIM + c;
#pragma unroll
  for (int i = 0; i < 4; ++i) unsafeAtomicAdd(ad + i, hs[i]);
}

__global__ __launch_bounds__(256) void input_proj(
    const float* __restrict__ x_occ, const float* __restrict__ cpo,
    const float* __restrict__ x_chord, const float* __restrict__ x_sec,
    const float* __restrict__ x_note, const float* __restrict__ x_sd,
    const float* __restrict__ W_occ, const float* __restrict__ b_occ,
    const float* __restrict__ W_chord, const float* __restrict__ b_chord,
    const float* __restrict__ W_sec, const float* __restrict__ b_sec,
    const float* __restrict__ W_note, const float* __restrict__ b_note,
    const float* __restrict__ W_sd, const float* __restrict__ b_sd,
    const float* __restrict__ type_emb, float* __restrict__ h) {
  size_t t = (size_t)blockIdx.x * 256 + threadIdx.x;
  if (t >= (size_t)NTOT * HDIM) return;
  int row = (int)(t >> 8);
  int c = (int)(t & 255);
  float acc;
  int type;
  if (row < NO_) {
    type = 0;
    const float* w = W_occ + (size_t)c * 96;
    const float* xo = x_occ + (size_t)row * 64;
    const float* xc = cpo + (size_t)row * 32;
    acc = b_occ[c];
    for (int k = 0; k < 64; ++k) acc += xo[k] * w[k];
    for (int k = 0; k < 32; ++k) acc += xc[k] * w[64 + k];
  } else if (row < NO_ + NC_) {
    type = 1;
    int r = row - NO_;
    const float* w = W_chord + (size_t)c * 32;
    const float* x = x_chord + (size_t)r * 32;
    acc = b_chord[c];
    for (int k = 0; k < 32; ++k) acc += x[k] * w[k];
  } else if (row < NO_ + NC_ + NS_) {
    type = 2;
    int r = row - (NO_ + NC_);
    const float* w = W_sec + (size_t)c * 16;
    const float* x = x_sec + (size_t)r * 16;
    acc = b_sec[c];
    for (int k = 0; k < 16; ++k) acc += x[k] * w[k];
  } else if (row < NO_ + NC_ + NS_ + NN_) {
    type = 3;
    int r = row - (NO_ + NC_ + NS_);
    const float* w = W_note + (size_t)c * 16;
    const float* x = x_note + (size_t)r * 16;
    acc = b_note[c];
    for (int k = 0; k < 16; ++k) acc += x[k] * w[k];
  } else {
    type = 4;
    int r = row - (NO_ + NC_ + NS_ + NN_);
    const float* w = W_sd + (size_t)c * 8;
    const float* x = x_sd + (size_t)r * 8;
    acc = b_sd[c];
    for (int k = 0; k < 8; ++k) acc += x[k] * w[k];
  }
  h[t] = acc + type_emb[type * HDIM + c];
}

__global__ void f32_to_f16(const float* __restrict__ s, _Float16* __restrict__ d,
                           size_t n) {
  size_t i = (size_t)blockIdx.x * blockDim.x + threadIdx.x;
  if (i < n) d[i] = (_Float16)s[i];
}

// h (f32, NTOT rows) -> f16 with zero padding up to MPAD rows
__global__ void h_to_f16(const float* __restrict__ h, _Float16* __restrict__ hh) {
  size_t i = (size_t)blockIdx.x * 256 + threadIdx.x;  // covers MPAD*HDIM
  size_t row = i >> 8;
  hh[i] = (row < NTOT) ? (_Float16)h[i] : (_Float16)0.f;
}

__global__ void agg_to_f16(const float* __restrict__ agg,
                           const float* __restrict__ invcnt,
                           _Float16* __restrict__ aggh) {
  size_t i = (size_t)blockIdx.x * 256 + threadIdx.x;
  size_t row = i >> 8;
  aggh[i] = (row < NTOT) ? (_Float16)(agg[i] * invcnt[row]) : (_Float16)0.f;
}

// LayerNorm(conv)*g + b + h  (wave32 per row, 8 cols/lane)
__global__ __launch_bounds__(256) void ln_residual(
    const float* __restrict__ conv, float* __restrict__ h,
    const float* __restrict__ g, const float* __restrict__ b) {
  int row = blockIdx.x * 8 + (threadIdx.x >> 5);
  int lane = threadIdx.x & 31;
  if (row >= NTOT) return;
  const float* cr = conv + (size_t)row * HDIM;
  float v[8];
  float s = 0.f;
#pragma unroll
  for (int i = 0; i < 8; ++i) { v[i] = cr[lane + 32 * i]; s += v[i]; }
#pragma unroll
  for (int o = 16; o > 0; o >>= 1) s += __shfl_xor(s, o, 32);
  float mu = s * (1.0f / HDIM);
  float q = 0.f;
#pragma unroll
  for (int i = 0; i < 8; ++i) { v[i] -= mu; q += v[i] * v[i]; }
#pragma unroll
  for (int o = 16; o > 0; o >>= 1) q += __shfl_xor(q, o, 32);
  float rstd = rsqrtf(q * (1.0f / HDIM) + LN_EPS);
  float* hr = h + (size_t)row * HDIM;
#pragma unroll
  for (int i = 0; i < 8; ++i) {
    int c = lane + 32 * i;
    hr[c] = v[i] * rstd * g[c] + b[c] + hr[c];
  }
}

// ---------------- launch ----------------------------------------------------
extern "C" void kernel_launch(void* const* d_in, const int* in_sizes, int n_in,
                              void* d_out, int out_size, void* d_ws, size_t ws_size,
                              hipStream_t stream) {
  (void)n_in; (void)out_size; (void)ws_size;
  const float* x_occ   = (const float*)d_in[0];
  const float* x_chord = (const float*)d_in[1];
  const float* x_sec   = (const float*)d_in[2];
  const float* x_note  = (const float*)d_in[3];
  const float* x_sd    = (const float*)d_in[4];
  const int* ei[10];
  int eN[10];
  for (int i = 0; i < 10; ++i) {
    ei[i] = (const int*)d_in[5 + i];
    eN[i] = in_sizes[5 + i] / 2;
  }
  const int srcOff[10] = {O_OCC, O_OCC, O_CHORD, O_OCC, O_SEC,
                          O_SEC, O_CHORD, O_NOTE, O_CHORD, O_SD};
  const int dstOff[10] = {O_OCC, O_CHORD, O_OCC, O_SEC, O_OCC,
                          O_SEC, O_NOTE, O_CHORD, O_SD, O_CHORD};
  const float* W_occ   = (const float*)d_in[15];
  const float* b_occ   = (const float*)d_in[16];
  const float* W_chord = (const float*)d_in[17];
  const float* b_chord = (const float*)d_in[18];
  const float* W_sec   = (const float*)d_in[19];
  const float* b_sec   = (const float*)d_in[20];
  const float* W_note  = (const float*)d_in[21];
  const float* b_note  = (const float*)d_in[22];
  const float* W_sd    = (const float*)d_in[23];
  const float* b_sd    = (const float*)d_in[24];
  const float* type_emb = (const float*)d_in[25];
  const float* Wl   = (const float*)d_in[26];
  const float* bl   = (const float*)d_in[27];
  const float* Wr   = (const float*)d_in[28];
  const float* ln_g = (const float*)d_in[29];
  const float* ln_b = (const float*)d_in[30];
  const float* Wc   = (const float*)d_in[31];
  const float* bc   = (const float*)d_in[32];
  float* out = (float*)d_out;

  // workspace bump allocator (256B aligned)
  char* p = (char*)d_ws;
  auto alloc = [&](size_t bytes) -> void* {
    void* r = (void*)p;
    p += (bytes + 255) & ~(size_t)255;
    return r;
  };
  float*    h      = (float*)alloc((size_t)MPAD * HDIM * 4);
  float*    agg    = (float*)alloc((size_t)MPAD * HDIM * 4);
  float*    conv   = (float*)alloc((size_t)MPAD * HDIM * 4);
  float*    invcnt = (float*)alloc((size_t)MPAD * 4);
  float*    cpo    = (float*)alloc((size_t)NO_ * 32 * 4);
  _Float16* hh     = (_Float16*)alloc((size_t)MPAD * HDIM * 2);
  _Float16* aggh   = (_Float16*)alloc((size_t)MPAD * HDIM * 2);
  _Float16* Wl16   = (_Float16*)alloc((size_t)NLAYER * HDIM * HDIM * 2);
  _Float16* Wr16   = (_Float16*)alloc((size_t)NLAYER * HDIM * HDIM * 2);
  _Float16* Wc16   = (_Float16*)alloc((size_t)NCLS * HDIM * 2);

  auto cdiv = [](long long a, long long b) { return (int)((a + b - 1) / b); };

  // chord features scattered onto occ nodes (ei_chord_occ = group 2)
  hipMemsetAsync(cpo, 0, (size_t)NO_ * 32 * 4, stream);
  hipMemsetAsync(invcnt, 0, (size_t)MPAD * 4, stream);
  scatter_chord<<<cdiv(eN[2], 256), 256, 0, stream>>>(x_chord, ei[2], eN[2], cpo);

  // in-degree -> inverse mean normalizer
  for (int g = 0; g < 10; ++g)
    count_deg<<<cdiv(eN[g], 256), 256, 0, stream>>>(ei[g], eN[g], dstOff[g], invcnt);
  inv_count<<<cdiv(NTOT, 256), 256, 0, stream>>>(invcnt);

  // type-specific input projections + type embedding
  input_proj<<<cdiv((long long)NTOT * HDIM, 256), 256, 0, stream>>>(
      x_occ, cpo, x_chord, x_sec, x_note, x_sd,
      W_occ, b_occ, W_chord, b_chord, W_sec, b_sec, W_note, b_note,
      W_sd, b_sd, type_emb, h);

  // f16 weight copies
  f32_to_f16<<<cdiv((long long)NLAYER * HDIM * HDIM, 256), 256, 0, stream>>>(
      Wl, Wl16, (size_t)NLAYER * HDIM * HDIM);
  f32_to_f16<<<cdiv((long long)NLAYER * HDIM * HDIM, 256), 256, 0, stream>>>(
      Wr, Wr16, (size_t)NLAYER * HDIM * HDIM);
  f32_to_f16<<<cdiv((long long)NCLS * HDIM, 256), 256, 0, stream>>>(
      Wc, Wc16, (size_t)NCLS * HDIM);

  const int mTiles = MPAD / 16;                       // 6597
  const int convBlocks = cdiv(mTiles, 8) * (HDIM / 64);   // 825 * 4
  for (int l = 0; l < NLAYER; ++l) {
    hipMemsetAsync(agg, 0, (size_t)MPAD * HDIM * 4, stream);
    for (int g = 0; g < 10; ++g)
      scatter_add<<<cdiv((long long)eN[g] * 64, 256), 256, 0, stream>>>(
          h, ei[g], eN[g], srcOff[g], dstOff[g], agg);
    h_to_f16<<<MPAD, 256, 0, stream>>>(h, hh);
    agg_to_f16<<<MPAD, 256, 0, stream>>>(agg, invcnt, aggh);
    gemm_dual_wmma<<<convBlocks, 256, 0, stream>>>(
        aggh, Wl16 + (size_t)l * HDIM * HDIM,
        hh,   Wr16 + (size_t)l * HDIM * HDIM,
        bl + (size_t)l * HDIM, conv, mTiles, HDIM, HDIM);
    ln_residual<<<cdiv(NTOT, 8), 256, 0, stream>>>(
        conv, h, ln_g + (size_t)l * HDIM, ln_b + (size_t)l * HDIM);
  }

  // classifier on occ nodes: out = h[:No] @ Wc^T + bc
  h_to_f16<<<MPAD, 256, 0, stream>>>(h, hh);
  const int clsBlocks = cdiv(NO_ / 16, 8) * (NCLS / 64);  // 625 * 16
  gemm_wmma<<<clsBlocks, 256, 0, stream>>>(
      hh, Wc16, bc, out, NO_ / 16, NCLS, HDIM);
}
